// MultiHeadAttention_20023137534362
// MI455X (gfx1250) — matrix operand order repaired and checked by reading, not compiled, not run on hardware
//
#include <hip/hip_runtime.h>
#include <hip/hip_bf16.h>
#include <stdint.h>

typedef __bf16 bf16x16 __attribute__((ext_vector_type(16)));
typedef float  floatx8 __attribute__((ext_vector_type(8)));
typedef int    v2i_t   __attribute__((vector_size(8)));   // async b64 payload type

#define DEV __device__ __forceinline__

static constexpr int Bb   = 4;
static constexpr int Ts   = 2048;
static constexpr int Cs   = 1024;
static constexpr int Hs   = 16;
static constexpr int Ds   = 64;
static constexpr int Ms   = Bb * Ts;   // 8192 rows (B*T)
static constexpr int Nqkv = 3 * Cs;    // 3072
static constexpr int Kd   = Cs;        // 1024

static constexpr int BMc = 128, BNc = 128, BKc = 32;
static constexpr int LDT = 36;         // GEMM LDS stride: 72B = 8B-aligned, 18-bank step (conflict-free)
static constexpr int QBLKc = 128, KBLKc = 64;
static constexpr int LDK = 68;         // attn LDS stride: 136B = 8B-aligned, 34-bank step (conflict-free)

// ---------------------------------------------------------------------------
// CDNA5 async global->LDS staging (ASYNCcnt-tracked), with sync fallback.
// ---------------------------------------------------------------------------
#define AS1 __attribute__((address_space(1)))
#define AS3 __attribute__((address_space(3)))

#if defined(__HIP_DEVICE_COMPILE__) && \
    __has_builtin(__builtin_amdgcn_global_load_async_to_lds_b64) && \
    __has_builtin(__builtin_amdgcn_s_wait_asynccnt)
#define USE_ASYNC 1
#else
#define USE_ASYNC 0
#endif

#if USE_ASYNC
DEV void cp_async8(__bf16* lds_dst, const __bf16* g_src) {
  // 8 bytes (4 bf16) per lane, DMA'd straight into LDS (no VGPR roundtrip).
  __builtin_amdgcn_global_load_async_to_lds_b64((AS1 v2i_t*)g_src,
                                                (AS3 v2i_t*)lds_dst, 0, 0);
}
#define ASYNC_WAIT_PREV() __builtin_amdgcn_s_wait_asynccnt(8)
#define ASYNC_WAIT_ALL()  __builtin_amdgcn_s_wait_asynccnt(0)
#else
#define ASYNC_WAIT_PREV()
#define ASYNC_WAIT_ALL()
#endif

DEV void copy8(__bf16* lds_dst, const __bf16* g_src) {
#if USE_ASYNC
  cp_async8(lds_dst, g_src);
#else
  *(uint2*)lds_dst = *(const uint2*)g_src;
#endif
}

// ---------------------------------------------------------------------------
// WMMA fragment loaders per CDNA5 ISA 7.12.2 (wave32).
//   A (16x32 bf16): lane&15 = M; VGPR v holds K = (v>>2)*16 + half*8 + (v&3)*2
//   B (32x16 bf16): lane&15 = N; VGPR v holds K as for A (lanes 0-15 K=0..7,16..23; lanes 16-31 K=8..15,24..31)
//   C/D (16x16 f32): lane&15 = N; VGPR r holds M = (lane>>4)*8 + r
// ---------------------------------------------------------------------------
DEV bf16x16 load_a_frag(const __bf16* tile, int stride, int kbase) {
  const int lane = threadIdx.x & 31;
  const int hlf  = lane >> 4;
  const int m    = lane & 15;
  bf16x16 f;
  #pragma unroll
  for (int v = 0; v < 8; ++v) {
    const int k = kbase + ((v >> 2) << 4) + (hlf << 3) + ((v & 3) << 1);
    const __bf16* p = tile + m * stride + k;
    f[2 * v]     = p[0];
    f[2 * v + 1] = p[1];
  }
  return f;
}

// tile stored [n][k] row-major
DEV bf16x16 load_b_frag_nmajor(const __bf16* tile, int stride, int kbase) {
  const int lane = threadIdx.x & 31;
  const int hlf  = lane >> 4;
  const int n    = lane & 15;
  bf16x16 f;
  #pragma unroll
  for (int v = 0; v < 8; ++v) {
    const int k = kbase + ((v >> 2) << 4) + (hlf << 3) + ((v & 3) << 1);
    const __bf16* p = tile + n * stride + k;
    f[2 * v]     = p[0];
    f[2 * v + 1] = p[1];
  }
  return f;
}

// tile stored [k][n] row-major
DEV bf16x16 load_b_frag_kmajor(const __bf16* tile, int stride, int kbase, int n0) {
  const int lane = threadIdx.x & 31;
  const int hlf  = lane >> 4;
  const int n    = n0 + (lane & 15);
  bf16x16 f;
  #pragma unroll
  for (int v = 0; v < 8; ++v) {
    const int k = kbase + ((v >> 2) << 4) + (hlf << 3) + ((v & 3) << 1);
    f[2 * v]     = tile[(k)     * stride + n];
    f[2 * v + 1] = tile[(k + 1) * stride + n];
  }
  return f;
}

DEV floatx8 wmma_bf16(bf16x16 a, bf16x16 b, floatx8 c) {
  return __builtin_amdgcn_wmma_f32_16x16x32_bf16(false, a, false, b,
                                                 (short)0, c, false, false);
}

// ---------------------------------------------------------------------------
// fp32 -> bf16 conversion
// ---------------------------------------------------------------------------
__global__ __launch_bounds__(256) void cvt_f32_bf16(const float* __restrict__ src,
                                                    __bf16* __restrict__ dst, int n) {
  const int i = blockIdx.x * 256 + threadIdx.x;
  if (i < n) dst[i] = (__bf16)src[i];
}

// ---------------------------------------------------------------------------
// Shared GEMM staging: one 128x32 A-tile and 128x32 W-tile (8 async b64/wave)
// ---------------------------------------------------------------------------
DEV void stage_gemm_tile(__bf16* dA, __bf16* dW,
                         const __bf16* A, const __bf16* W,
                         int m0, int n0, int k0, int tid) {
  for (int c = tid; c < (BMc * BKc / 4); c += 256) {   // 1024 chunks of 4 bf16
    const int r = c >> 3, kc = (c & 7) << 2;
    copy8(dA + r * LDT + kc, A + (size_t)(m0 + r) * Kd + (k0 + kc));
    copy8(dW + r * LDT + kc, W + (size_t)(n0 + r) * Kd + (k0 + kc));
  }
}

// ---------------------------------------------------------------------------
// QKV GEMM: qkv[M,3C] = x[M,K] * W[3C,K]^T + b; scatter to Q/K/V [B,H,T,D] bf16
// 128x128 block, BK=32, double-buffered async LDS; 8 waves 2(M)x4(N), 4x2 wmma
// ---------------------------------------------------------------------------
__global__ __launch_bounds__(256) void qkv_gemm_kernel(
    const __bf16* __restrict__ A, const __bf16* __restrict__ W,
    const float* __restrict__ bias,
    __bf16* __restrict__ Qo, __bf16* __restrict__ Ko, __bf16* __restrict__ Vo) {
  __shared__ __bf16 As[2][BMc * LDT];
  __shared__ __bf16 Ws[2][BNc * LDT];
  const int tid  = threadIdx.x;
  const int lane = tid & 31, wid = tid >> 5;
  const int wm = wid >> 2, wn = wid & 3;
  const int m0 = blockIdx.x * BMc, n0 = blockIdx.y * BNc;

  floatx8 zero8;
  #pragma unroll
  for (int i = 0; i < 8; ++i) zero8[i] = 0.0f;
  floatx8 acc[4][2];
  #pragma unroll
  for (int mt = 0; mt < 4; ++mt)
    #pragma unroll
    for (int nt = 0; nt < 2; ++nt) acc[mt][nt] = zero8;

  constexpr int NKT = Kd / BKc;   // 32
  stage_gemm_tile(As[0], Ws[0], A, W, m0, n0, 0, tid);

  for (int kt = 0; kt < NKT; ++kt) {
    const int cur = kt & 1;
    if (kt + 1 < NKT) {
      stage_gemm_tile(As[cur ^ 1], Ws[cur ^ 1], A, W, m0, n0, (kt + 1) * BKc, tid);
      ASYNC_WAIT_PREV();          // stage kt landed; stage kt+1 still in flight
    } else {
      ASYNC_WAIT_ALL();
    }
    __syncthreads();

    bf16x16 af[4], bfr[2];
    #pragma unroll
    for (int mt = 0; mt < 4; ++mt)
      af[mt] = load_a_frag(&As[cur][(wm * 64 + mt * 16) * LDT], LDT, 0);
    #pragma unroll
    for (int nt = 0; nt < 2; ++nt)
      bfr[nt] = load_b_frag_nmajor(&Ws[cur][(wn * 32 + nt * 16) * LDT], LDT, 0);
    #pragma unroll
    for (int mt = 0; mt < 4; ++mt)
      #pragma unroll
      for (int nt = 0; nt < 2; ++nt)
        acc[mt][nt] = wmma_bf16(af[mt], bfr[nt], acc[mt][nt]);

    __syncthreads();              // done reading buf[cur] before it is re-staged
  }

  const int hlf = lane >> 4, nc = lane & 15;
  #pragma unroll
  for (int mt = 0; mt < 4; ++mt) {
    #pragma unroll
    for (int nt = 0; nt < 2; ++nt) {
      const int gn = n0 + wn * 32 + nt * 16 + nc;     // 0..3071
      const float bv = bias[gn];
      const int which = gn >> 10;                     // 0=Q 1=K 2=V
      const int nn = gn & (Cs - 1);
      const int h = nn >> 6, d = nn & (Ds - 1);
      __bf16* outp = (which == 0) ? Qo : ((which == 1) ? Ko : Vo);
      #pragma unroll
      for (int r = 0; r < 8; ++r) {
        const int gm = m0 + wm * 64 + mt * 16 + (hlf << 3) + r;
        const int t = gm & (Ts - 1);
        const int b = gm >> 11;
        const float v = acc[mt][nt][r] + bv;
        outp[((size_t)(b * Hs + h) * Ts + t) * Ds + d] = (__bf16)v;
      }
    }
  }
}

// ---------------------------------------------------------------------------
// Attention K/V staging: one 64x64 K-tile + V-tile (8 async b64 per wave)
// ---------------------------------------------------------------------------
DEV void stage_kv(__bf16* dK, __bf16* dV, const __bf16* Kt, const __bf16* V,
                  size_t head, int kb, int tid) {
  for (int c = tid; c < (KBLKc * Ds / 4); c += 256) {  // 1024 chunks of 4 bf16
    const int r = c >> 4, kc = (c & 15) << 2;
    const size_t g = head + (size_t)(kb * KBLKc + r) * Ds + kc;
    copy8(dK + r * LDK + kc, Kt + g);
    copy8(dV + r * LDK + kc, V + g);
  }
}

// ---------------------------------------------------------------------------
// Flash attention: block = (head, 128 q-rows); 8 waves x 16 rows.
// Double-buffered async K/V streaming, causal (2*qb+2 blocks), online softmax.
// ---------------------------------------------------------------------------
__global__ __launch_bounds__(256) void attn_kernel(
    const __bf16* __restrict__ Q, const __bf16* __restrict__ Kt,
    const __bf16* __restrict__ V, __bf16* __restrict__ Yo) {
  __shared__ __bf16 Ks[2][KBLKc * LDK];
  __shared__ __bf16 Vs[2][KBLKc * LDK];
  __shared__ __bf16 Ps[8 * 16 * LDK];             // per-wave P relayout buffer
  const int tid = threadIdx.x, lane = tid & 31, wid = tid >> 5;
  const int hlf = lane >> 4, nc = lane & 15;
  const int bh = blockIdx.x;
  const int b = bh >> 4, h = bh & 15;
  const int q0 = blockIdx.y * QBLKc;
  const size_t head = (size_t)bh * Ts * Ds;

  // Q as A-fragments (rows q0 + wid*16 + 0..15, K-dim D=64 -> 2 frags)
  bf16x16 qf[2];
  {
    const int qrow = q0 + wid * 16 + nc;
    const __bf16* qp = Q + head + (size_t)qrow * Ds;
    #pragma unroll
    for (int f = 0; f < 2; ++f)
      #pragma unroll
      for (int v = 0; v < 8; ++v) {
        const int k = f * 32 + ((v >> 2) << 4) + (hlf << 3) + ((v & 3) << 1);
        qf[f][2 * v]     = qp[k];
        qf[f][2 * v + 1] = qp[k + 1];
      }
  }

  floatx8 o[4];
  float mrow[8], lrow[8];
  #pragma unroll
  for (int r = 0; r < 8; ++r) { mrow[r] = -__builtin_inff(); lrow[r] = 0.0f; }
  #pragma unroll
  for (int dt = 0; dt < 4; ++dt)
    #pragma unroll
    for (int r = 0; r < 8; ++r) o[dt][r] = 0.0f;

  const int nkb = (q0 + QBLKc) / KBLKc;   // causal block count
  stage_kv(Ks[0], Vs[0], Kt, V, head, 0, tid);

  for (int kb = 0; kb < nkb; ++kb) {
    const int cur = kb & 1;
    if (kb + 1 < nkb) {
      stage_kv(Ks[cur ^ 1], Vs[cur ^ 1], Kt, V, head, kb + 1, tid);
      ASYNC_WAIT_PREV();
    } else {
      ASYNC_WAIT_ALL();
    }
    __syncthreads();

    // S = Q K^T (16 rows x 64 cols per wave)
    float s[4][8];
    #pragma unroll
    for (int nt = 0; nt < 4; ++nt) {
      floatx8 sa;
      #pragma unroll
      for (int r = 0; r < 8; ++r) sa[r] = 0.0f;
      #pragma unroll
      for (int ks = 0; ks < 2; ++ks) {
        bf16x16 kf = load_b_frag_nmajor(&Ks[cur][(nt * 16) * LDK], LDK, ks * 32);
        sa = wmma_bf16(qf[ks], kf, sa);
      }
      #pragma unroll
      for (int r = 0; r < 8; ++r) s[nt][r] = sa[r];
    }

    // scale + causal mask + per-row block max
    float bm[8];
    #pragma unroll
    for (int r = 0; r < 8; ++r) bm[r] = -__builtin_inff();
    #pragma unroll
    for (int nt = 0; nt < 4; ++nt) {
      const int col = kb * KBLKc + nt * 16 + nc;
      #pragma unroll
      for (int r = 0; r < 8; ++r) {
        const int row = q0 + wid * 16 + (hlf << 3) + r;
        float sv = s[nt][r] * 0.125f;               // 1/sqrt(64)
        if (col > row) sv = -__builtin_inff();
        s[nt][r] = sv;
        bm[r] = fmaxf(bm[r], sv);
      }
    }
    #pragma unroll
    for (int msk = 1; msk < 16; msk <<= 1)          // stays within 16-lane half
      #pragma unroll
      for (int r = 0; r < 8; ++r)
        bm[r] = fmaxf(bm[r], __shfl_xor(bm[r], msk, 32));

    // online softmax update
    float alpha[8], rs[8];
    #pragma unroll
    for (int r = 0; r < 8; ++r) {
      const float mn = fmaxf(mrow[r], bm[r]);
      alpha[r] = __expf(mrow[r] - mn);
      mrow[r] = mn;
      rs[r] = 0.0f;
    }
    #pragma unroll
    for (int nt = 0; nt < 4; ++nt)
      #pragma unroll
      for (int r = 0; r < 8; ++r) {
        const float p = __expf(s[nt][r] - mrow[r]);
        s[nt][r] = p;
        rs[r] += p;
      }
    #pragma unroll
    for (int msk = 1; msk < 16; msk <<= 1)
      #pragma unroll
      for (int r = 0; r < 8; ++r)
        rs[r] += __shfl_xor(rs[r], msk, 32);
    #pragma unroll
    for (int r = 0; r < 8; ++r) lrow[r] = lrow[r] * alpha[r] + rs[r];
    #pragma unroll
    for (int dt = 0; dt < 4; ++dt)
      #pragma unroll
      for (int r = 0; r < 8; ++r) o[dt][r] *= alpha[r];

    // P: C/D layout -> per-wave LDS row-major -> A fragments
    __bf16* pw = &Ps[wid * 16 * LDK];
    #pragma unroll
    for (int nt = 0; nt < 4; ++nt)
      #pragma unroll
      for (int r = 0; r < 8; ++r)
        pw[((hlf << 3) + r) * LDK + nt * 16 + nc] = (__bf16)s[nt][r];
    asm volatile("s_wait_dscnt 0" ::: "memory");    // per-wave LDS RAW fence

    // O += P V
    #pragma unroll
    for (int ks = 0; ks < 2; ++ks) {
      bf16x16 pa = load_a_frag(pw, LDK, ks * 32);
      #pragma unroll
      for (int dt = 0; dt < 4; ++dt) {
        bf16x16 vf = load_b_frag_kmajor(Vs[cur], LDK, ks * 32, dt * 16);
        o[dt] = wmma_bf16(pa, vf, o[dt]);
      }
    }

    __syncthreads();              // done with buf[cur] before it is re-staged
  }

  // normalize and write y [B,T,C] bf16
  #pragma unroll
  for (int dt = 0; dt < 4; ++dt)
    #pragma unroll
    for (int r = 0; r < 8; ++r) {
      const int t = q0 + wid * 16 + (hlf << 3) + r;
      const float val = o[dt][r] / lrow[r];
      Yo[(size_t)(b * Ts + t) * Cs + h * Ds + dt * 16 + nc] = (__bf16)val;
    }
}

// ---------------------------------------------------------------------------
// Output projection: out[M,C] = y[M,C] * W_proj[C,C]^T + b_proj (fp32 out)
// ---------------------------------------------------------------------------
__global__ __launch_bounds__(256) void proj_gemm_kernel(
    const __bf16* __restrict__ A, const __bf16* __restrict__ W,
    const float* __restrict__ bias, float* __restrict__ out) {
  __shared__ __bf16 As[2][BMc * LDT];
  __shared__ __bf16 Ws[2][BNc * LDT];
  const int tid  = threadIdx.x;
  const int lane = tid & 31, wid = tid >> 5;
  const int wm = wid >> 2, wn = wid & 3;
  const int m0 = blockIdx.x * BMc, n0 = blockIdx.y * BNc;

  floatx8 zero8;
  #pragma unroll
  for (int i = 0; i < 8; ++i) zero8[i] = 0.0f;
  floatx8 acc[4][2];
  #pragma unroll
  for (int mt = 0; mt < 4; ++mt)
    #pragma unroll
    for (int nt = 0; nt < 2; ++nt) acc[mt][nt] = zero8;

  constexpr int NKT = Kd / BKc;
  stage_gemm_tile(As[0], Ws[0], A, W, m0, n0, 0, tid);

  for (int kt = 0; kt < NKT; ++kt) {
    const int cur = kt & 1;
    if (kt + 1 < NKT) {
      stage_gemm_tile(As[cur ^ 1], Ws[cur ^ 1], A, W, m0, n0, (kt + 1) * BKc, tid);
      ASYNC_WAIT_PREV();
    } else {
      ASYNC_WAIT_ALL();
    }
    __syncthreads();

    bf16x16 af[4], bfr[2];
    #pragma unroll
    for (int mt = 0; mt < 4; ++mt)
      af[mt] = load_a_frag(&As[cur][(wm * 64 + mt * 16) * LDT], LDT, 0);
    #pragma unroll
    for (int nt = 0; nt < 2; ++nt)
      bfr[nt] = load_b_frag_nmajor(&Ws[cur][(wn * 32 + nt * 16) * LDT], LDT, 0);
    #pragma unroll
    for (int mt = 0; mt < 4; ++mt)
      #pragma unroll
      for (int nt = 0; nt < 2; ++nt)
        acc[mt][nt] = wmma_bf16(af[mt], bfr[nt], acc[mt][nt]);

    __syncthreads();
  }

  const int hlf = lane >> 4, nc = lane & 15;
  #pragma unroll
  for (int mt = 0; mt < 4; ++mt)
    #pragma unroll
    for (int nt = 0; nt < 2; ++nt) {
      const int gn = n0 + wn * 32 + nt * 16 + nc;
      const float bv = bias[gn];
      #pragma unroll
      for (int r = 0; r < 8; ++r) {
        const int gm = m0 + wm * 64 + mt * 16 + (hlf << 3) + r;
        out[(size_t)gm * Cs + gn] = acc[mt][nt][r] + bv;
      }
    }
}

// ---------------------------------------------------------------------------
// Host launcher
// ---------------------------------------------------------------------------
extern "C" void kernel_launch(void* const* d_in, const int* in_sizes, int n_in,
                              void* d_out, int out_size, void* d_ws, size_t ws_size,
                              hipStream_t stream) {
  const float* x     = (const float*)d_in[0];
  const float* Wqkv  = (const float*)d_in[1];
  const float* bqkv  = (const float*)d_in[2];
  const float* Wproj = (const float*)d_in[3];
  const float* bproj = (const float*)d_in[4];
  (void)in_sizes; (void)n_in; (void)out_size; (void)ws_size;

  char* ws = (char*)d_ws;
  size_t off = 0;
  auto take = [&](size_t bytes) -> void* {
    void* p = ws + off;
    off += (bytes + 255) & ~(size_t)255;
    return p;
  };
  const size_t NX  = (size_t)Ms * Kd;     // 8,388,608 (== B*T*C == B*H*T*D)
  const size_t NWQ = (size_t)Nqkv * Kd;   // 3,145,728
  const size_t NWP = (size_t)Cs * Cs;     // 1,048,576

  __bf16* xb  = (__bf16*)take(NX * 2);
  __bf16* wqb = (__bf16*)take(NWQ * 2);
  __bf16* wpb = (__bf16*)take(NWP * 2);
  __bf16* Qb  = (__bf16*)take(NX * 2);
  __bf16* Kb  = (__bf16*)take(NX * 2);
  __bf16* Vb  = (__bf16*)take(NX * 2);
  __bf16* yb  = (__bf16*)take(NX * 2);    // ~88 MB of d_ws total

  cvt_f32_bf16<<<dim3((unsigned)((NX  + 255) / 256)), dim3(256), 0, stream>>>(x, xb, (int)NX);
  cvt_f32_bf16<<<dim3((unsigned)((NWQ + 255) / 256)), dim3(256), 0, stream>>>(Wqkv, wqb, (int)NWQ);
  cvt_f32_bf16<<<dim3((unsigned)((NWP + 255) / 256)), dim3(256), 0, stream>>>(Wproj, wpb, (int)NWP);

  qkv_gemm_kernel<<<dim3(Ms / 128, Nqkv / 128), dim3(256), 0, stream>>>(
      xb, wqb, bqkv, Qb, Kb, Vb);
  attn_kernel<<<dim3(Bb * Hs, Ts / 128), dim3(256), 0, stream>>>(Qb, Kb, Vb, yb);
  proj_gemm_kernel<<<dim3(Ms / 128, Cs / 128), dim3(256), 0, stream>>>(
      yb, wpb, bproj, (float*)d_out);
}
